// GPT2_64123861729815
// MI455X (gfx1250) — matrix operand order repaired and checked by reading, not compiled, not run on hardware
//
#include <hip/hip_runtime.h>

// GPT-2 forward for gfx1250 (MI455X). bf16 WMMA + TDM async tile staging.
#define LNUM 12
#define HNUM 12
#define EDIM 768
#define VDIM 50257
#define DH   64
#define BDIM 2
#define TDIM 1024
#define MROWS (BDIM * TDIM)   // 2048

typedef __bf16 bf16_t;
typedef __attribute__((ext_vector_type(16))) __bf16 v16bf;
typedef __attribute__((ext_vector_type(8)))  float  v8f;
typedef __attribute__((ext_vector_type(4)))  unsigned int v4u;
typedef __attribute__((ext_vector_type(8)))  int v8i;
typedef __attribute__((ext_vector_type(4)))  int v4i;

static __device__ __forceinline__ float gelu_f(float x) {
    const float c = 0.7978845608028654f;  // sqrt(2/pi)
    float x3 = x * x * x;
    return 0.5f * x * (1.0f + tanhf(c * (x + 0.044715f * x3)));
}

// Issue one TDM 2D tile load (bf16 elements) into LDS.
// tile: tile_k (dim0, contiguous) x tile_n (dim1 rows); row stride = stride_elems.
// tensor dims = remaining extent from tile origin (OOB rows/cols read as zero).
static __device__ __forceinline__ void tdm_load_2d_bf16(
        unsigned lds_off, const void* gaddr,
        unsigned tile_k, unsigned tile_n,
        unsigned long long stride_elems,
        unsigned long long tdim0, unsigned long long tdim1) {
    unsigned long long ga = (unsigned long long)gaddr;
    v4u g0;
    g0[0] = 1u;                                   // count=1, user descriptor
    g0[1] = lds_off;                              // LDS byte address
    g0[2] = (unsigned)ga;                         // global_addr[31:0]
    g0[3] = (unsigned)((ga >> 32) & 0x01FFFFFFull) | 0x80000000u;  // addr[56:32] | type=2
    v8i g1;
    g1[0] = (int)(1u << 16);                      // workgroup_mask=0, data_size=1 (2B)
    g1[1] = (int)((unsigned)(tdim0 & 0xFFFFu) << 16);              // tensor_dim0[15:0]
    g1[2] = (int)(((tdim0 >> 16) & 0xFFFFu) | ((tdim1 & 0xFFFFu) << 16));
    g1[3] = (int)(((tdim1 >> 16) & 0xFFFFu) | (tile_k << 16));     // tile_dim0
    g1[4] = (int)tile_n;                          // tile_dim1 (tile_dim2=0)
    g1[5] = (int)(unsigned)(stride_elems & 0xFFFFFFFFull);         // dim0_stride[31:0]
    g1[6] = (int)(unsigned)((stride_elems >> 32) & 0xFFFFull);     // dim0_stride[47:32]
    g1[7] = 0;
    v4i z4 = {0, 0, 0, 0};
    v8i z8 = {0, 0, 0, 0, 0, 0, 0, 0};
    __builtin_amdgcn_tensor_load_to_lds(g0, g1, z4, z4, z8, 0);
}

// ---------------------------------------------------------------- embeddings
__global__ void gpt2_embed_kernel(const int* __restrict__ ids,
                                  const float* __restrict__ wte,
                                  const float* __restrict__ wpe,
                                  float* __restrict__ h) {
    size_t idx = (size_t)blockIdx.x * blockDim.x + threadIdx.x;
    if (idx >= (size_t)MROWS * EDIM) return;
    int m = (int)(idx / EDIM), e = (int)(idx % EDIM);
    int t = m % TDIM;
    int id = ids[m];
    h[idx] = wte[(size_t)id * EDIM + e] + wpe[(size_t)t * EDIM + e];
}

// ---------------------------------------------------------------- mask output
__global__ void gpt2_mask_kernel(const int* __restrict__ am, float* __restrict__ out) {
    size_t idx = (size_t)blockIdx.x * blockDim.x + threadIdx.x;
    if (idx >= (size_t)BDIM * TDIM * TDIM) return;
    int b = (int)(idx / ((size_t)TDIM * TDIM));
    int r = (int)(idx % ((size_t)TDIM * TDIM));
    int q = r / TDIM, k = r % TDIM;
    float v = (k <= q) ? 1.0f : 0.0f;
    v *= (float)am[b * TDIM + k] * (float)am[b * TDIM + q];
    out[idx] = v;
}

// ---------------------------------------------------------------- layernorm -> bf16
__launch_bounds__(256)
__global__ void gpt2_ln_kernel(const float* __restrict__ x,
                               const float* __restrict__ g,
                               const float* __restrict__ bb,
                               bf16_t* __restrict__ out) {
    __shared__ float red[16];
    int row = blockIdx.x;
    const float* xr = x + (size_t)row * EDIM;
    float s = 0.f, s2 = 0.f;
    for (int i = threadIdx.x; i < EDIM; i += 256) {
        float v = xr[i]; s += v; s2 += v * v;
    }
    for (int off = 16; off >= 1; off >>= 1) {
        s  += __shfl_xor(s,  off, 32);
        s2 += __shfl_xor(s2, off, 32);
    }
    int wv = threadIdx.x >> 5;
    if ((threadIdx.x & 31) == 0) { red[wv] = s; red[8 + wv] = s2; }
    __syncthreads();
    float ts = 0.f, ts2 = 0.f;
    for (int i = 0; i < 8; ++i) { ts += red[i]; ts2 += red[8 + i]; }
    float mu = ts * (1.0f / EDIM);
    float var = ts2 * (1.0f / EDIM) - mu * mu;
    float rs = rsqrtf(var + 1e-5f);
    for (int i = threadIdx.x; i < EDIM; i += 256)
        out[(size_t)row * EDIM + i] = (bf16_t)((xr[i] - mu) * rs * g[i] + bb[i]);
}

// ---------------------------------------------------------------- weight convert
// src [K,N] f32 (or [N,K] if tsrc) -> dst [N,K] bf16
__global__ void gpt2_wconv_kernel(const float* __restrict__ W, bf16_t* __restrict__ out,
                                  int Ndim, int Kdim, int tsrc) {
    size_t idx = (size_t)blockIdx.x * blockDim.x + threadIdx.x;
    if (idx >= (size_t)Ndim * Kdim) return;
    int n = (int)(idx / Kdim), k = (int)(idx % Kdim);
    float v = tsrc ? W[idx] : W[(size_t)k * Ndim + n];
    out[idx] = (bf16_t)v;
}

// ---------------------------------------------------------------- qkv pack (bf16, V transposed)
__global__ void gpt2_pack_qkv_kernel(const float* __restrict__ qkv,
                                     bf16_t* __restrict__ qb,
                                     bf16_t* __restrict__ kb,
                                     bf16_t* __restrict__ vt) {
    size_t idx = (size_t)blockIdx.x * blockDim.x + threadIdx.x;
    if (idx >= (size_t)MROWS * 3 * EDIM) return;
    int m = (int)(idx / (3 * EDIM)), c = (int)(idx % (3 * EDIM));
    int b = m / TDIM, t = m % TDIM;
    int which = c / EDIM, e = c % EDIM;
    int hh = e / DH, d = e % DH;
    bf16_t v = (bf16_t)qkv[idx];
    if (which == 0)      qb[(((size_t)b * HNUM + hh) * TDIM + t) * DH + d] = v;
    else if (which == 1) kb[(((size_t)b * HNUM + hh) * TDIM + t) * DH + d] = v;
    else                 vt[(((size_t)b * HNUM + hh) * DH + d) * TDIM + t] = v;
}

// ---------------------------------------------------------------- WMMA GEMM (TDM-staged)
// C[M,N] = A(bf16 [M,K]) x Wt(bf16 [N,K]); double-buffered TDM tiles in LDS.
// flags: 1 = GELU epilogue
#define BMT 128
#define BNT 128
#define BKT 32
__launch_bounds__(256)
__global__ void gpt2_gemm_kernel(const bf16_t* __restrict__ A,
                                 const bf16_t* __restrict__ Wt,
                                 const float* __restrict__ bias,
                                 const float* __restrict__ resid,
                                 float* __restrict__ Cf,
                                 bf16_t* __restrict__ Cb,
                                 int Ndim, int Kdim, int flags) {
    __shared__ bf16_t sA[2][BMT][BKT];   // 16 KB
    __shared__ bf16_t sB[2][BNT][BKT];   // 16 KB, [n][k]
    const int tid  = threadIdx.x;
    const int lane = tid & 31;
    const int wave = tid >> 5;        // 8 waves: 2 (M) x 4 (N)
    const int wm   = wave >> 2;
    const int wn   = wave & 3;
    const int l16  = lane & 15;
    const int hl   = lane >> 4;
    const int m0 = blockIdx.y * BMT;
    const int n0 = blockIdx.x * BNT;
    const int nk = Kdim / BKT;

    v8f acc[4][2] = {};

    // prologue: stage k-tile 0
    if (wave == 0) {
        tdm_load_2d_bf16((unsigned)(size_t)&sA[0][0][0], A + (size_t)m0 * Kdim,
                         BKT, BMT, (unsigned long long)Kdim,
                         (unsigned long long)Kdim, (unsigned long long)(MROWS - m0));
        tdm_load_2d_bf16((unsigned)(size_t)&sB[0][0][0], Wt + (size_t)n0 * Kdim,
                         BKT, BNT, (unsigned long long)Kdim,
                         (unsigned long long)Kdim, (unsigned long long)(Ndim - n0));
    }

    for (int ki = 0; ki < nk; ++ki) {
        int cur = ki & 1;
        if (wave == 0) {
            if (ki + 1 < nk) {
                int k1 = (ki + 1) * BKT;
                tdm_load_2d_bf16((unsigned)(size_t)&sA[cur ^ 1][0][0],
                                 A + (size_t)m0 * Kdim + k1,
                                 BKT, BMT, (unsigned long long)Kdim,
                                 (unsigned long long)(Kdim - k1),
                                 (unsigned long long)(MROWS - m0));
                tdm_load_2d_bf16((unsigned)(size_t)&sB[cur ^ 1][0][0],
                                 Wt + (size_t)n0 * Kdim + k1,
                                 BKT, BNT, (unsigned long long)Kdim,
                                 (unsigned long long)(Kdim - k1),
                                 (unsigned long long)(Ndim - n0));
                __builtin_amdgcn_s_wait_tensorcnt(2);  // in-order: tile ki landed
            } else {
                __builtin_amdgcn_s_wait_tensorcnt(0);
            }
        }
        __syncthreads();   // tile ki visible to all waves

        // fragments (ISA 16-bit layouts) and WMMA
        v16bf af[4], bfr[2];
        for (int mt = 0; mt < 4; ++mt) {
            const bf16_t* p = &sA[cur][wm * 64 + mt * 16 + l16][0];
            union { v16bf v; uint4 u[2]; } t;
            t.u[0] = *(const uint4*)(p + 8 * hl);        // elems 0..7 : K = 8h+e
            t.u[1] = *(const uint4*)(p + 16 + 8 * hl);   // elems 8..15: K = 16+8h+e
            af[mt] = t.v;
        }
        for (int nt = 0; nt < 2; ++nt) {
            const bf16_t* p = &sB[cur][wn * 32 + nt * 16 + l16][0];
            union { v16bf v; uint4 u[2]; } t;
            t.u[0] = *(const uint4*)(p + 8 * hl);        // elems 0..7 : K = 8h+e
            t.u[1] = *(const uint4*)(p + 16 + 8 * hl);   // elems 8..15: K = 16+8h+e
            bfr[nt] = t.v;
        }
        for (int mt = 0; mt < 4; ++mt)
            for (int nt = 0; nt < 2; ++nt)
                acc[mt][nt] = __builtin_amdgcn_wmma_f32_16x16x32_bf16(
                    false, af[mt], false, bfr[nt], (short)0, acc[mt][nt],
                    false, false);
        __syncthreads();   // done reading buf[cur]; safe to overwrite next iter
    }

    // ---- epilogue
    for (int mt = 0; mt < 4; ++mt)
        for (int nt = 0; nt < 2; ++nt) {
            int n = n0 + wn * 32 + nt * 16 + l16;
            bool nok = (n < Ndim);
            float bv = (bias != nullptr && nok) ? bias[n] : 0.f;
            for (int r = 0; r < 8; ++r) {
                int m = m0 + wm * 64 + mt * 16 + r + 8 * hl;
                float v = acc[mt][nt][r] + bv;
                if (resid != nullptr && nok) v += resid[(size_t)m * Ndim + n];
                if (flags & 1) v = gelu_f(v);
                if (nok) {
                    if (Cf != nullptr) Cf[(size_t)m * Ndim + n] = v;
                    else               Cb[(size_t)m * Ndim + n] = (bf16_t)v;
                }
            }
        }
}

// ---------------------------------------------------------------- flash attention (WMMA)
// grid: (T/64, H, B); block: 128 (4 waves); each wave owns a 16-row Q tile.
__launch_bounds__(128)
__global__ void gpt2_attn_kernel(const bf16_t* __restrict__ qb,
                                 const bf16_t* __restrict__ kb,
                                 const bf16_t* __restrict__ vt,
                                 const int* __restrict__ am,
                                 bf16_t* __restrict__ y) {
    __shared__ bf16_t sP[4][16][32];  // per-wave P staging (4 KB)
    const int qt = blockIdx.x, hh = blockIdx.y, b = blockIdx.z;
    const int tid = threadIdx.x, lane = tid & 31, wave = tid >> 5;
    const int l16 = lane & 15, hl = lane >> 4;
    const int qrow0 = qt * 64 + wave * 16;
    const size_t bh = (size_t)b * HNUM + hh;
    const bf16_t* qptr = qb + bh * TDIM * DH;   // [T, D]
    const bf16_t* kptr = kb + bh * TDIM * DH;   // [T, D]
    const bf16_t* vptr = vt + bh * DH * TDIM;   // [D, T]

    v16bf qf[2];
    {
        const bf16_t* p = qptr + (size_t)(qrow0 + l16) * DH;
        for (int kk = 0; kk < 2; ++kk) {
            union { v16bf v; uint4 u[2]; } t;
            t.u[0] = *(const uint4*)(p + kk * 32 + 8 * hl);
            t.u[1] = *(const uint4*)(p + kk * 32 + 16 + 8 * hl);
            qf[kk] = t.v;
        }
    }

    v8f o[4] = {};
    float mrow[8], lrow[8];
    for (int r = 0; r < 8; ++r) { mrow[r] = -3.0e38f; lrow[r] = 0.f; }

    const int kend = qrow0 + 16;  // causal bound for this wave
    for (int kb0 = 0; kb0 < kend; kb0 += 32) {
        v8f s[2] = {};
        for (int nt = 0; nt < 2; ++nt) {
            const bf16_t* p = kptr + (size_t)(kb0 + nt * 16 + l16) * DH;
            for (int kk = 0; kk < 2; ++kk) {
                union { v16bf v; uint4 u[2]; } t;
                t.u[0] = *(const uint4*)(p + kk * 32 + 8 * hl);
                t.u[1] = *(const uint4*)(p + kk * 32 + 16 + 8 * hl);
                s[nt] = __builtin_amdgcn_wmma_f32_16x16x32_bf16(
                    false, qf[kk], false, t.v, (short)0, s[nt], false, false);
            }
        }
        const float scale = 0.125f;  // 1/sqrt(64)
        for (int nt = 0; nt < 2; ++nt) {
            int key = kb0 + nt * 16 + l16;
            float amk = (float)am[b * TDIM + key];
            for (int r = 0; r < 8; ++r) {
                int q = qrow0 + r + 8 * hl;
                float amq = (float)am[b * TDIM + q];
                float v = s[nt][r] * scale;
                if (key > q || amk == 0.f || amq == 0.f) v = -1.0e20f;
                s[nt][r] = v;
            }
        }
        for (int r = 0; r < 8; ++r) {
            float mx = fmaxf(s[0][r], s[1][r]);
            for (int off = 8; off >= 1; off >>= 1) mx = fmaxf(mx, __shfl_xor(mx, off, 16));
            float mnew = fmaxf(mrow[r], mx);
            float alpha = __expf(mrow[r] - mnew);
            float p0 = __expf(s[0][r] - mnew);
            float p1 = __expf(s[1][r] - mnew);
            float sum = p0 + p1;
            for (int off = 8; off >= 1; off >>= 1) sum += __shfl_xor(sum, off, 16);
            lrow[r] = lrow[r] * alpha + sum;
            mrow[r] = mnew;
            s[0][r] = p0; s[1][r] = p1;
            for (int nt = 0; nt < 4; ++nt) o[nt][r] *= alpha;
        }
        for (int nt = 0; nt < 2; ++nt)
            for (int r = 0; r < 8; ++r)
                sP[wave][r + 8 * hl][nt * 16 + l16] = (bf16_t)s[nt][r];
        asm volatile("s_wait_dscnt 0" ::: "memory");
        v16bf pf;
        {
            const bf16_t* p = &sP[wave][l16][0];
            union { v16bf v; uint4 u[2]; } t;
            t.u[0] = *(const uint4*)(p + 8 * hl);
            t.u[1] = *(const uint4*)(p + 16 + 8 * hl);
            pf = t.v;
        }
        for (int nt = 0; nt < 4; ++nt) {
            const bf16_t* p = vptr + (size_t)(nt * 16 + l16) * TDIM + kb0;
            union { v16bf v; uint4 u[2]; } t;
            t.u[0] = *(const uint4*)(p + 8 * hl);
            t.u[1] = *(const uint4*)(p + 16 + 8 * hl);
            o[nt] = __builtin_amdgcn_wmma_f32_16x16x32_bf16(
                false, pf, false, t.v, (short)0, o[nt], false, false);
        }
    }

    for (int r = 0; r < 8; ++r) {
        float inv = 1.0f / lrow[r];
        int q = qrow0 + r + 8 * hl;
        for (int nt = 0; nt < 4; ++nt) {
            int d = nt * 16 + l16;
            y[((size_t)b * TDIM + q) * EDIM + hh * DH + d] = (bf16_t)(o[nt][r] * inv);
        }
    }
}

// ---------------------------------------------------------------- host orchestration
extern "C" void kernel_launch(void* const* d_in, const int* in_sizes, int n_in,
                              void* d_out, int out_size, void* d_ws, size_t ws_size,
                              hipStream_t stream) {
    const int*   ids    = (const int*)d_in[0];
    const int*   am     = (const int*)d_in[1];
    const float* wte    = (const float*)d_in[2];
    const float* wpe    = (const float*)d_in[3];
    const float* ln1_g  = (const float*)d_in[4];
    const float* ln1_b  = (const float*)d_in[5];
    const float* attn_w = (const float*)d_in[6];
    const float* attn_b = (const float*)d_in[7];
    const float* proj_w = (const float*)d_in[8];
    const float* proj_b = (const float*)d_in[9];
    const float* ln2_g  = (const float*)d_in[10];
    const float* ln2_b  = (const float*)d_in[11];
    const float* fc_w   = (const float*)d_in[12];
    const float* fc_b   = (const float*)d_in[13];
    const float* fc2_w  = (const float*)d_in[14];
    const float* fc2_b  = (const float*)d_in[15];
    const float* lnf_g  = (const float*)d_in[16];
    const float* lnf_b  = (const float*)d_in[17];
    const float* lm_w   = (const float*)d_in[18];
    float* out = (float*)d_out;

    char* ws = (char*)d_ws;
    size_t off = 0;
    auto wsa = [&](size_t bytes) -> void* {
        void* p = ws + off;
        off += (bytes + 255) & ~(size_t)255;
        return p;
    };
    float*  h    = (float*) wsa((size_t)MROWS * EDIM * 4);
    bf16_t* xn   = (bf16_t*)wsa((size_t)MROWS * EDIM * 2);
    float*  qkv  = (float*) wsa((size_t)MROWS * 3 * EDIM * 4);
    bf16_t* qb   = (bf16_t*)wsa((size_t)MROWS * EDIM * 2);
    bf16_t* kbuf = (bf16_t*)wsa((size_t)MROWS * EDIM * 2);
    bf16_t* vt   = (bf16_t*)wsa((size_t)MROWS * EDIM * 2);
    bf16_t* yb   = (bf16_t*)wsa((size_t)MROWS * EDIM * 2);
    bf16_t* ub   = (bf16_t*)wsa((size_t)MROWS * 4 * EDIM * 2);
    bf16_t* hn   = (bf16_t*)wsa((size_t)MROWS * EDIM * 2);
    bf16_t* wbf  = (bf16_t*)wsa((size_t)VDIM * EDIM * 2);   // shared [N,K] bf16 weight buf
    (void)ws_size; (void)in_sizes; (void)n_in; (void)out_size;

    dim3 blk256(256), blk128(128);
    auto cdiv = [](size_t a, size_t b) -> unsigned { return (unsigned)((a + b - 1) / b); };

    gpt2_embed_kernel<<<cdiv((size_t)MROWS * EDIM, 256), blk256, 0, stream>>>(ids, wte, wpe, h);
    gpt2_mask_kernel<<<cdiv((size_t)BDIM * TDIM * TDIM, 256), blk256, 0, stream>>>(
        am, out + (size_t)BDIM * TDIM * VDIM);

    for (int l = 0; l < LNUM; ++l) {
        // --- attention block
        gpt2_ln_kernel<<<MROWS, blk256, 0, stream>>>(h, ln1_g + l * EDIM, ln1_b + l * EDIM, xn);
        gpt2_wconv_kernel<<<cdiv((size_t)3 * EDIM * EDIM, 256), blk256, 0, stream>>>(
            attn_w + (size_t)l * EDIM * 3 * EDIM, wbf, 3 * EDIM, EDIM, 0);
        gpt2_gemm_kernel<<<dim3(3 * EDIM / BNT, MROWS / BMT), blk256, 0, stream>>>(
            xn, wbf, attn_b + (size_t)l * 3 * EDIM, nullptr, qkv, nullptr, 3 * EDIM, EDIM, 0);
        gpt2_pack_qkv_kernel<<<cdiv((size_t)MROWS * 3 * EDIM, 256), blk256, 0, stream>>>(
            qkv, qb, kbuf, vt);
        gpt2_attn_kernel<<<dim3(TDIM / 64, HNUM, BDIM), blk128, 0, stream>>>(qb, kbuf, vt, am, yb);
        gpt2_wconv_kernel<<<cdiv((size_t)EDIM * EDIM, 256), blk256, 0, stream>>>(
            proj_w + (size_t)l * EDIM * EDIM, wbf, EDIM, EDIM, 0);
        gpt2_gemm_kernel<<<dim3(EDIM / BNT, MROWS / BMT), blk256, 0, stream>>>(
            yb, wbf, proj_b + (size_t)l * EDIM, /*resid=*/h, /*Cf=*/h, nullptr, EDIM, EDIM, 0);
        // --- MLP block
        gpt2_ln_kernel<<<MROWS, blk256, 0, stream>>>(h, ln2_g + l * EDIM, ln2_b + l * EDIM, xn);
        gpt2_wconv_kernel<<<cdiv((size_t)4 * EDIM * EDIM, 256), blk256, 0, stream>>>(
            fc_w + (size_t)l * EDIM * 4 * EDIM, wbf, 4 * EDIM, EDIM, 0);
        gpt2_gemm_kernel<<<dim3(4 * EDIM / BNT, MROWS / BMT), blk256, 0, stream>>>(
            xn, wbf, fc_b + (size_t)l * 4 * EDIM, nullptr, nullptr, ub, 4 * EDIM, EDIM, /*GELU*/1);
        gpt2_wconv_kernel<<<cdiv((size_t)4 * EDIM * EDIM, 256), blk256, 0, stream>>>(
            fc2_w + (size_t)l * 4 * EDIM * EDIM, wbf, EDIM, 4 * EDIM, 0);
        gpt2_gemm_kernel<<<dim3(EDIM / BNT, MROWS / BMT), blk256, 0, stream>>>(
            ub, wbf, fc2_b + (size_t)l * EDIM, /*resid=*/h, /*Cf=*/h, nullptr, EDIM, 4 * EDIM, 0);
    }

    gpt2_ln_kernel<<<MROWS, blk256, 0, stream>>>(h, lnf_g, lnf_b, hn);
    gpt2_wconv_kernel<<<cdiv((size_t)VDIM * EDIM, 256), blk256, 0, stream>>>(
        lm_w, wbf, VDIM, EDIM, /*already [N,K]*/1);
    gpt2_gemm_kernel<<<dim3((VDIM + BNT - 1) / BNT, MROWS / BMT), blk256, 0, stream>>>(
        hn, wbf, nullptr, nullptr, out, nullptr, VDIM, EDIM, 0);
}
